// WinLayer_55903294324770
// MI455X (gfx1250) — matrix operand order repaired and checked by reading, not compiled, not run on hardware
//
#include <hip/hip_runtime.h>
#include <stdint.h>

// ---------------- problem constants ----------------
#define NB      8
#define C_DIM   256
#define IMG     112
#define IMG2    12544      // IMG*IMG
#define MW      7
#define WIN     16
#define T_TOK   256        // WIN*WIN
#define HEADS   4
#define DH      64
#define NWIN    392        // NB*MW*MW
#define C3      768
#define LDA     264        // padded bf16 row stride for 256-wide tiles

typedef unsigned int  u32;
typedef unsigned short u16;
typedef unsigned long long u64;
typedef __attribute__((ext_vector_type(16))) __bf16 v16bf;
typedef __attribute__((ext_vector_type(8)))  float  v8f;
typedef __attribute__((ext_vector_type(4)))  float  f32x4;
typedef __attribute__((ext_vector_type(4)))  u32    u32x4;

// ---------------- scalar helpers ----------------
__device__ __forceinline__ u16 f2bf(float f) {
  u32 u = __builtin_bit_cast(u32, f);
  u32 r = (u + 0x7fffu + ((u >> 16) & 1u)) >> 16;   // round-to-nearest-even
  return (u16)r;
}
__device__ __forceinline__ float bf2f(u16 h) {
  u32 u = ((u32)h) << 16;
  return __builtin_bit_cast(float, u);
}
__device__ __forceinline__ float gelu_tanh(float x) {
  float x3 = x * x * x;
  return 0.5f * x * (1.f + tanhf(0.7978845608028654f * (x + 0.044715f * x3)));
}

// ---------------- CDNA5 async memory->LDS copy (ASYNCcnt path) ----------------
// GLOBAL_LOAD_ASYNC_TO_LDS_B128: per-lane 16B global -> LDS, no VGPR data,
// tracked with ASYNCcnt (CDNA5 ISA). LDS operand is the
// low 32 bits of the generic shared-memory pointer (ISA: LDS generic addresses
// truncate to the byte offset).
__device__ __forceinline__ void async_copy16(const void* sptr, const void* gaddr) {
  u32 lds = (u32)(uintptr_t)sptr;
  u64 ga  = (u64)(uintptr_t)gaddr;
  asm volatile("global_load_async_to_lds_b128 %0, %1, off"
               :: "v"(lds), "v"(ga) : "memory");
}
__device__ __forceinline__ void wait_async0() {
  asm volatile("s_wait_asynccnt 0" ::: "memory");
}

// ---------------- WMMA fragment helpers ----------------
// A fragment: 16x32 bf16 tile from row-major LDS [M][K] (stride lda elems).
// ISA layout (16-bit A 16x32): lane m=lane&15, hi=lane>>4;
//   VGPR0-3 halves = K: k0+8*hi .. +7 ; VGPR4-7 halves = K: k0+16+8*hi .. +7
__device__ __forceinline__ v16bf load_a(const u16* As, int lda, int row0, int k0) {
  int lane = threadIdx.x & 31;
  int row  = row0 + (lane & 15);
  int hi   = lane >> 4;
  const u32x4* p0 = (const u32x4*)(As + row * lda + k0 + 8 * hi);
  const u32x4* p1 = (const u32x4*)(As + row * lda + k0 + 16 + 8 * hi);
  union { u32x4 u[2]; v16bf v; } f;
  f.u[0] = *p0;
  f.u[1] = *p1;
  return f.v;
}
// B fragment: 32x16 bf16 tile from transposed LDS Bt[N][K] (stride ldb elems).
// lane n=lane&15, hi=lane>>4; halves 0-7: K=k0+8*hi..+7 ; halves 8-15: K=k0+16+8*hi..+7
__device__ __forceinline__ v16bf load_b(const u16* Bt, int ldb, int col0, int k0) {
  int lane = threadIdx.x & 31;
  int col  = col0 + (lane & 15);
  int hi   = lane >> 4;
  const u32x4* p0 = (const u32x4*)(Bt + col * ldb + k0 + 8 * hi);
  const u32x4* p1 = (const u32x4*)(Bt + col * ldb + k0 + 16 + 8 * hi);
  union { u32x4 u[2]; v16bf v; } f;
  f.u[0] = *p0;
  f.u[1] = *p1;
  return f.v;
}
__device__ __forceinline__ v8f wmma_bf16(v16bf a, v16bf b, v8f c) {
  return __builtin_amdgcn_wmma_f32_16x16x32_bf16(false, a, false, b, (short)0, c,
                                                 false, false);
}

// ---------------- K1: window gather + LayerNorm1 -> bf16 hA ----------------
__global__ __launch_bounds__(256) void k_ln1(const float* __restrict__ X,
                                             const float* __restrict__ g,
                                             const float* __restrict__ be,
                                             u16* __restrict__ hA) {
  int b = blockIdx.x;
  int n = b / 49, rem = b % 49, my = rem / 7, mx = rem % 7;
  int t = threadIdx.x;
  int hh = my * WIN + (t >> 4), ww = mx * WIN + (t & 15);
  const float* base = X + (size_t)n * C_DIM * IMG2 + hh * IMG + ww;
  float s = 0.f, ss = 0.f;
  for (int c = 0; c < C_DIM; ++c) {
    float v = base[(size_t)c * IMG2];
    s += v; ss += v * v;
  }
  float mu = s * (1.f / 256.f);
  float var = ss * (1.f / 256.f) - mu * mu;
  float rs = rsqrtf(var + 1e-5f);
  u16* dst = hA + ((size_t)b * T_TOK + t) * C_DIM;
  for (int c0 = 0; c0 < C_DIM; c0 += 8) {
    u32x4 pk;
    #pragma unroll
    for (int w = 0; w < 4; ++w) {
      int c = c0 + 2 * w;
      float v0 = (base[(size_t)c * IMG2] - mu) * rs * g[c] + be[c];
      float v1 = (base[(size_t)(c + 1) * IMG2] - mu) * rs * g[c + 1] + be[c + 1];
      pk[w] = (u32)f2bf(v0) | ((u32)f2bf(v1) << 16);
    }
    *(u32x4*)(dst + c0) = pk;
  }
}

// ---------------- K2: QKV GEMM (per window, 128-col group) ----------------
// C[256x128] = hA[256x256] @ Wqkv[:,grp*128..]; q section scaled by 1/sqrt(DH).
__global__ __launch_bounds__(256) void k_qkv(const u16* __restrict__ hA,
                                             const float* __restrict__ Wqkv,
                                             const float* __restrict__ bqkv,
                                             u16* __restrict__ qkv) {
  __shared__ __align__(16) u16 As[T_TOK * LDA];   // 135168 B
  __shared__ __align__(16) u16 Bt[128 * LDA];     //  67584 B
  int b = blockIdx.x, grp = blockIdx.y, tid = threadIdx.x;
  {  // A: async copy of this thread's token row (512 B) straight into LDS
    const u16* src = hA + ((size_t)b * T_TOK + tid) * C_DIM;
    u16* dst = As + tid * LDA;
    #pragma unroll
    for (int i = 0; i < 32; ++i) async_copy16(dst + 8 * i, src + 8 * i);
  }
  {  // B: coalesced read of Wqkv[k][grp*128+n], store transposed bf16 (paired)
    int nn = tid & 127;
    int kb = (tid >> 7) * 128;
    for (int k = 0; k < 128; k += 2) {
      u32 p = (u32)f2bf(Wqkv[(size_t)(kb + k) * C3 + grp * 128 + nn]) |
              ((u32)f2bf(Wqkv[(size_t)(kb + k + 1) * C3 + grp * 128 + nn]) << 16);
      *(u32*)(Bt + nn * LDA + kb + k) = p;
    }
  }
  wait_async0();
  __syncthreads();
  int wave = tid >> 5, lane = tid & 31, nl = lane & 15, hi = lane >> 4;
  int m0w = (wave >> 1) * 64, n0w = (wave & 1) * 64;
  v8f acc[4][4] = {};
  for (int k0 = 0; k0 < 256; k0 += 32) {
    v16bf a[4], bb[4];
    #pragma unroll
    for (int i = 0; i < 4; ++i) a[i] = load_a(As, LDA, m0w + 16 * i, k0);
    #pragma unroll
    for (int j = 0; j < 4; ++j) bb[j] = load_b(Bt, LDA, n0w + 16 * j, k0);
    #pragma unroll
    for (int i = 0; i < 4; ++i)
      #pragma unroll
      for (int j = 0; j < 4; ++j) acc[i][j] = wmma_bf16(a[i], bb[j], acc[i][j]);
  }
  #pragma unroll
  for (int i = 0; i < 4; ++i)
    #pragma unroll
    for (int j = 0; j < 4; ++j) {
      v8f a = acc[i][j];
      int col = grp * 128 + n0w + 16 * j + nl;
      float scale = (col < 256) ? 0.125f : 1.f;   // fold 1/sqrt(64) into q
      float bias = bqkv[col];
      #pragma unroll
      for (int r = 0; r < 8; ++r) {
        int m = m0w + 16 * i + r + 8 * hi;
        qkv[((size_t)b * T_TOK + m) * C3 + col] = f2bf((a[r] + bias) * scale);
      }
    }
}

// ---------------- K3: attention per (window, head) ----------------
__global__ __launch_bounds__(256) void k_attn(const u16* __restrict__ qkv,
                                              u16* __restrict__ o) {
  __shared__ __align__(16) u16 qs[T_TOK * 72];    // q [t][d]
  __shared__ __align__(16) u16 ks[T_TOK * 72];    // k [t][d]
  __shared__ __align__(16) u16 vt[DH * LDA];      // v^T [d][t]
  __shared__ __align__(16) u16 sb[T_TOK * LDA];   // scores / attn [t][s]
  __shared__ float rinv[T_TOK];
  int b = blockIdx.x, hd = blockIdx.y, t = threadIdx.x;
  const u16* rowq = qkv + ((size_t)b * T_TOK + t) * C3 + hd * DH;
  {  // q,k rows: async global->LDS (row-major, already A/B layout)
    #pragma unroll
    for (int i = 0; i < 8; ++i) {
      async_copy16(qs + t * 72 + 8 * i, rowq + 8 * i);
      async_copy16(ks + t * 72 + 8 * i, rowq + 256 + 8 * i);
    }
    // v needs a transpose -> register path
    const u32x4* sv = (const u32x4*)(rowq + 512);
    #pragma unroll
    for (int i = 0; i < 8; ++i) {
      u32x4 vv = sv[i];
      #pragma unroll
      for (int w = 0; w < 4; ++w) {
        int d = i * 8 + 2 * w;
        vt[d * LDA + t] = (u16)(vv[w] & 0xffffu);
        vt[(d + 1) * LDA + t] = (u16)(vv[w] >> 16);
      }
    }
  }
  wait_async0();
  __syncthreads();
  int wave = t >> 5, lane = t & 31, nl = lane & 15, hi = lane >> 4;
  // scores S = q @ k^T  (M=256, N=256 in two halves, K=64)
  for (int nh = 0; nh < 2; ++nh) {
    int m0w = (wave >> 1) * 64, n0w = nh * 128 + (wave & 1) * 64;
    v8f acc[4][4] = {};
    for (int k0 = 0; k0 < 64; k0 += 32) {
      v16bf a[4], bb[4];
      #pragma unroll
      for (int i = 0; i < 4; ++i) a[i] = load_a(qs, 72, m0w + 16 * i, k0);
      #pragma unroll
      for (int j = 0; j < 4; ++j) bb[j] = load_b(ks, 72, n0w + 16 * j, k0);
      #pragma unroll
      for (int i = 0; i < 4; ++i)
        #pragma unroll
        for (int j = 0; j < 4; ++j) acc[i][j] = wmma_bf16(a[i], bb[j], acc[i][j]);
    }
    #pragma unroll
    for (int i = 0; i < 4; ++i)
      #pragma unroll
      for (int j = 0; j < 4; ++j) {
        v8f a = acc[i][j];
        #pragma unroll
        for (int r = 0; r < 8; ++r)
          sb[(m0w + 16 * i + r + 8 * hi) * LDA + n0w + 16 * j + nl] = f2bf(a[r]);
      }
  }
  __syncthreads();
  {  // softmax over row t, vectorized b128 LDS traffic; 1/sum deferred to PV
    u32x4* row4 = (u32x4*)(sb + t * LDA);
    float mx = -3.0e38f;
    #pragma unroll 4
    for (int i = 0; i < 32; ++i) {
      u32x4 pk = row4[i];
      #pragma unroll
      for (int w = 0; w < 4; ++w) {
        mx = fmaxf(mx, bf2f((u16)(pk[w] & 0xffffu)));
        mx = fmaxf(mx, bf2f((u16)(pk[w] >> 16)));
      }
    }
    float sum = 0.f;
    #pragma unroll 2
    for (int i = 0; i < 32; ++i) {
      u32x4 pk = row4[i];
      u32x4 out;
      #pragma unroll
      for (int w = 0; w < 4; ++w) {
        float e0 = __expf(bf2f((u16)(pk[w] & 0xffffu)) - mx);
        float e1 = __expf(bf2f((u16)(pk[w] >> 16)) - mx);
        sum += e0 + e1;
        out[w] = (u32)f2bf(e0) | ((u32)f2bf(e1) << 16);
      }
      row4[i] = out;
    }
    rinv[t] = 1.f / sum;
  }
  __syncthreads();
  {  // O = attn @ v  (M=256, N=64, K=256); waves split M 8 ways
    int m0w = wave * 32;
    v8f acc[2][4] = {};
    for (int k0 = 0; k0 < 256; k0 += 32) {
      v16bf a[2], bb[4];
      #pragma unroll
      for (int i = 0; i < 2; ++i) a[i] = load_a(sb, LDA, m0w + 16 * i, k0);
      #pragma unroll
      for (int j = 0; j < 4; ++j) bb[j] = load_b(vt, LDA, 16 * j, k0);
      #pragma unroll
      for (int i = 0; i < 2; ++i)
        #pragma unroll
        for (int j = 0; j < 4; ++j) acc[i][j] = wmma_bf16(a[i], bb[j], acc[i][j]);
    }
    #pragma unroll
    for (int i = 0; i < 2; ++i)
      #pragma unroll
      for (int j = 0; j < 4; ++j) {
        v8f a = acc[i][j];
        #pragma unroll
        for (int r = 0; r < 8; ++r) {
          int m = m0w + 16 * i + r + 8 * hi;
          o[((size_t)b * T_TOK + m) * C_DIM + hd * DH + 16 * j + nl] =
              f2bf(a[r] * rinv[m]);
        }
      }
  }
}

// ---------------- shared 256x256x256 GEMM body ----------------
__device__ __forceinline__ void load_AB(const u16* __restrict__ Aglob,
                                        const float* __restrict__ W,
                                        u16* As, u16* Bt) {
  int tid = threadIdx.x;
  {  // A: async copy of this thread's row (512 B) straight into LDS
    const u16* src = Aglob + (size_t)tid * C_DIM;
    u16* dst = As + tid * LDA;
    #pragma unroll
    for (int i = 0; i < 32; ++i) async_copy16(dst + 8 * i, src + 8 * i);
  }
  for (int k = 0; k < 256; k += 2) {              // coalesced over tid (=col)
    u32 p = (u32)f2bf(W[(size_t)k * C_DIM + tid]) |
            ((u32)f2bf(W[(size_t)(k + 1) * C_DIM + tid]) << 16);
    *(u32*)(Bt + tid * LDA + k) = p;
  }
  wait_async0();
}

template <typename Epi>
__device__ __forceinline__ void gemm_window(const u16* As, const u16* Bt, Epi&& epi) {
  int tid = threadIdx.x;
  int wave = tid >> 5, lane = tid & 31, nl = lane & 15, hi = lane >> 4;
  for (int nh = 0; nh < 2; ++nh) {
    int m0w = (wave >> 1) * 64, n0w = nh * 128 + (wave & 1) * 64;
    v8f acc[4][4] = {};
    for (int k0 = 0; k0 < 256; k0 += 32) {
      v16bf a[4], bb[4];
      #pragma unroll
      for (int i = 0; i < 4; ++i) a[i] = load_a(As, LDA, m0w + 16 * i, k0);
      #pragma unroll
      for (int j = 0; j < 4; ++j) bb[j] = load_b(Bt, LDA, n0w + 16 * j, k0);
      #pragma unroll
      for (int i = 0; i < 4; ++i)
        #pragma unroll
        for (int j = 0; j < 4; ++j) acc[i][j] = wmma_bf16(a[i], bb[j], acc[i][j]);
    }
    #pragma unroll
    for (int i = 0; i < 4; ++i)
      #pragma unroll
      for (int j = 0; j < 4; ++j) {
        v8f a = acc[i][j];
        #pragma unroll
        for (int r = 0; r < 8; ++r)
          epi(m0w + 16 * i + r + 8 * hi, n0w + 16 * j + nl, a[r]);
      }
  }
}

// ---------------- K4a: o @ Wo + bo + gather(X) -> x2 (f32) ----------------
__global__ __launch_bounds__(256) void k_oproj(const u16* __restrict__ o,
                                               const float* __restrict__ Wo,
                                               const float* __restrict__ bo,
                                               const float* __restrict__ X,
                                               float* __restrict__ x2) {
  __shared__ __align__(16) u16 As[T_TOK * LDA];
  __shared__ __align__(16) u16 Bt[C_DIM * LDA];
  int b = blockIdx.x;
  int n = b / 49, rem = b % 49, my = rem / 7, mx = rem % 7;
  load_AB(o + (size_t)b * T_TOK * C_DIM, Wo, As, Bt);
  __syncthreads();
  const float* xb = X + (size_t)n * C_DIM * IMG2;
  float* xrow = x2 + (size_t)b * T_TOK * C_DIM;
  gemm_window(As, Bt, [&](int m, int c, float acc) {
    int hh = my * WIN + (m >> 4), ww = mx * WIN + (m & 15);
    float xg = xb[(size_t)c * IMG2 + hh * IMG + ww];
    xrow[(size_t)m * C_DIM + c] = acc + bo[c] + xg;
  });
}

// ---------------- K4b: LayerNorm2 over x2 rows -> bf16 h2 ----------------
__global__ __launch_bounds__(256) void k_ln2(const float* __restrict__ x2,
                                             const float* __restrict__ g,
                                             const float* __restrict__ be,
                                             u16* __restrict__ h2) {
  int b = blockIdx.x, t = threadIdx.x;
  const float* row = x2 + ((size_t)b * T_TOK + t) * C_DIM;
  const f32x4* row4 = (const f32x4*)row;
  float s = 0.f, ss = 0.f;
  #pragma unroll 4
  for (int i = 0; i < 64; ++i) {
    f32x4 v = row4[i];
    #pragma unroll
    for (int w = 0; w < 4; ++w) { s += v[w]; ss += v[w] * v[w]; }
  }
  float mu = s * (1.f / 256.f);
  float var = ss * (1.f / 256.f) - mu * mu;
  float rs = rsqrtf(var + 1e-5f);
  u16* dst = h2 + ((size_t)b * T_TOK + t) * C_DIM;
  for (int c0 = 0; c0 < C_DIM; c0 += 8) {
    u32x4 pk;
    #pragma unroll
    for (int w = 0; w < 4; ++w) {
      int c = c0 + 2 * w;
      float v0 = (row[c] - mu) * rs * g[c] + be[c];
      float v1 = (row[c + 1] - mu) * rs * g[c + 1] + be[c + 1];
      pk[w] = (u32)f2bf(v0) | ((u32)f2bf(v1) << 16);
    }
    *(u32x4*)(dst + c0) = pk;
  }
}

// ---------------- K5: gelu(h2 @ W1 + b1) -> bf16 gws ----------------
__global__ __launch_bounds__(256) void k_mlp1(const u16* __restrict__ h2,
                                              const float* __restrict__ W1,
                                              const float* __restrict__ b1,
                                              u16* __restrict__ gws) {
  __shared__ __align__(16) u16 As[T_TOK * LDA];
  __shared__ __align__(16) u16 Bt[C_DIM * LDA];
  int b = blockIdx.x;
  load_AB(h2 + (size_t)b * T_TOK * C_DIM, W1, As, Bt);
  __syncthreads();
  u16* grow = gws + (size_t)b * T_TOK * C_DIM;
  gemm_window(As, Bt, [&](int m, int c, float acc) {
    grow[(size_t)m * C_DIM + c] = f2bf(gelu_tanh(acc + b1[c]));
  });
}

// ---------------- K6: gws @ W2 + b2 + x2 -> scatter NCHW out ----------------
__global__ __launch_bounds__(256) void k_mlp2(const u16* __restrict__ gws,
                                              const float* __restrict__ W2,
                                              const float* __restrict__ b2,
                                              const float* __restrict__ x2,
                                              float* __restrict__ out) {
  __shared__ __align__(16) u16 As[T_TOK * LDA];
  __shared__ __align__(16) u16 Bt[C_DIM * LDA];
  int b = blockIdx.x;
  int n = b / 49, rem = b % 49, my = rem / 7, mx = rem % 7;
  load_AB(gws + (size_t)b * T_TOK * C_DIM, W2, As, Bt);
  __syncthreads();
  const float* xrow = x2 + (size_t)b * T_TOK * C_DIM;
  float* ob = out + (size_t)n * C_DIM * IMG2;
  gemm_window(As, Bt, [&](int m, int c, float acc) {
    int hh = my * WIN + (m >> 4), ww = mx * WIN + (m & 15);
    ob[(size_t)c * IMG2 + hh * IMG + ww] =
        acc + b2[c] + xrow[(size_t)m * C_DIM + c];
  });
}

// ---------------- host launcher ----------------
extern "C" void kernel_launch(void* const* d_in, const int* in_sizes, int n_in,
                              void* d_out, int out_size, void* d_ws, size_t ws_size,
                              hipStream_t stream) {
  (void)in_sizes; (void)n_in; (void)out_size; (void)ws_size;
  const float* X    = (const float*)d_in[0];
  const float* Wqkv = (const float*)d_in[1];
  const float* bqkv = (const float*)d_in[2];
  const float* Wo   = (const float*)d_in[3];
  const float* bo   = (const float*)d_in[4];
  const float* W1   = (const float*)d_in[5];
  const float* b1   = (const float*)d_in[6];
  const float* W2   = (const float*)d_in[7];
  const float* b2   = (const float*)d_in[8];
  const float* ln1g = (const float*)d_in[9];
  const float* ln1b = (const float*)d_in[10];
  const float* ln2g = (const float*)d_in[11];
  const float* ln2b = (const float*)d_in[12];

  // workspace layout (regions reused across pipeline stages):
  //   [0, SZ_QKV)           : qkv bf16 (K2->K3), then x2 f32 (K4a->K6)
  //   [SZ_QKV, +SZ_BF)      : hA bf16 (K1->K2), then h2 bf16 (K4b->K5)
  //   [SZ_QKV+SZ_BF, +SZ_BF): o bf16 (K3->K4a), then gelu bf16 (K5->K6)
  const size_t SZ_BF  = (size_t)NWIN * T_TOK * C_DIM * sizeof(u16);      //  51.4 MB
  const size_t SZ_QKV = (size_t)NWIN * T_TOK * C3 * sizeof(u16);         // 154.1 MB
  char* w = (char*)d_ws;
  u16*   qkv = (u16*)w;
  float* x2  = (float*)w;                    // overlays qkv after K3 completes
  u16*   hA  = (u16*)(w + SZ_QKV);
  u16*   hB  = (u16*)(w + SZ_QKV + SZ_BF);

  k_ln1  <<<dim3(NWIN),       256, 0, stream>>>(X, ln1g, ln1b, hA);
  k_qkv  <<<dim3(NWIN, 6),    256, 0, stream>>>(hA, Wqkv, bqkv, qkv);
  k_attn <<<dim3(NWIN, HEADS),256, 0, stream>>>(qkv, hB);
  k_oproj<<<dim3(NWIN),       256, 0, stream>>>(hB, Wo, bo, X, x2);
  k_ln2  <<<dim3(NWIN),       256, 0, stream>>>(x2, ln2g, ln2b, hA);
  k_mlp1 <<<dim3(NWIN),       256, 0, stream>>>(hA, W1, b1, hB);
  k_mlp2 <<<dim3(NWIN),       256, 0, stream>>>(hB, W2, b2, x2, (float*)d_out);
}
